// HypPostEnc_37838661878284
// MI455X (gfx1250) — matrix operand order repaired and checked by reading, not compiled, not run on hardware
//
#include <hip/hip_runtime.h>
#include <hip/hip_bf16.h>

typedef __attribute__((ext_vector_type(16))) _Float16 v16h;
typedef __attribute__((ext_vector_type(8)))  _Float16 v8h;
typedef __attribute__((ext_vector_type(8)))  float    v8f;
typedef __attribute__((ext_vector_type(4)))  unsigned int u32x4;
typedef __attribute__((ext_vector_type(8)))  int      i32x8;
typedef __attribute__((ext_vector_type(4)))  int      i32x4;

#define TT    64
#define BB    1024
#define EE    300
#define EPAD  320
#define HH    256
#define G3    768
#define DD    512
#define NROWS 65536      // T*B
#define EPS_  1e-7f

__device__ __forceinline__ float sigm(float x){ return 1.0f/(1.0f + __expf(-x)); }

// A fragment (16x32 f16) from an LDS tile with given row stride (halfs).
// Lanes 0-15: row=lane, K = {0..15}; lanes 16-31: row=lane-16, K = {16..31}.
__device__ __forceinline__ v16h load_a_frag(const _Float16* lds, int stride, int kc, int lane){
  const int row  = lane & 15;
  const int half = lane >> 4;
  const _Float16* p = lds + row*stride + kc*32 + half*16;
  v8h lo = *(const v8h*)p;
  v8h hi = *(const v8h*)(p + 8);
  v16h a;
#pragma unroll
  for (int i = 0; i < 8; ++i){ a[i] = lo[i]; a[i+8] = hi[i]; }
  return a;
}

// B fragment (32x16 f16), stored pre-shuffled: frag base + lane*16 halfs contiguous.
__device__ __forceinline__ v16h load_b_frag(const _Float16* g, int frag, int lane){
  return *(const v16h*)(g + ((size_t)frag*32 + (size_t)lane)*16);
}

__device__ __forceinline__ v8f wmma_f16(v16h a, v16h b, v8f c){
  return __builtin_amdgcn_wmma_f32_16x16x32_f16(false, a, false, b, (short)0, c, false, false);
}

// ---------------------------------------------------------------------------
// Tensor Data Mover helpers: build D# per CDNA5 ISA.
// 2D tile: addr = global + data_size*(x + y*stride0); tile = tile_d0 x tile_d1.
// ---------------------------------------------------------------------------
__device__ __forceinline__ void tdm_load_2d(unsigned lds_addr, unsigned long long gaddr,
                                            unsigned dsz_code, unsigned tensor_d0,
                                            unsigned tensor_d1, unsigned tile_d0,
                                            unsigned tile_d1, unsigned long long stride0)
{
  u32x4 g0;
  g0[0] = 1u;                                   // count=1, no gather
  g0[1] = lds_addr;                             // LDS byte offset
  g0[2] = (unsigned)gaddr;                      // global_addr[31:0]
  g0[3] = ((unsigned)((gaddr >> 32) & 0x01FFFFFFull)) | (2u << 30);  // [56:32] | type=2
  i32x8 g1;
  g1[0] = (int)(dsz_code << 16);                // wg_mask=0 | data_size
  g1[1] = (int)((tensor_d0 & 0xFFFFu) << 16);   // tensor_dim0[15:0]
  g1[2] = (int)(((tensor_d0 >> 16) & 0xFFFFu) | ((tensor_d1 & 0xFFFFu) << 16));
  g1[3] = (int)(((tensor_d1 >> 16) & 0xFFFFu) | (tile_d0 << 16));
  g1[4] = (int)(tile_d1 & 0xFFFFu);             // tile_dim1 | tile_dim2=0
  g1[5] = (int)(unsigned)(stride0 & 0xFFFFFFFFull);
  g1[6] = (int)(unsigned)((stride0 >> 32) & 0xFFFFull);
  g1[7] = 0;
  i32x4 z4; z4[0]=0; z4[1]=0; z4[2]=0; z4[3]=0;
#if __clang_major__ >= 23
  i32x8 z8; z8[0]=0; z8[1]=0; z8[2]=0; z8[3]=0; z8[4]=0; z8[5]=0; z8[6]=0; z8[7]=0;
  __builtin_amdgcn_tensor_load_to_lds(g0, g1, z4, z4, z8, 0);
#else
  __builtin_amdgcn_tensor_load_to_lds(g0, g1, z4, z4, 0);
#endif
}

// Gather mode: 16 x 16-bit row indices in groups 2/3; each row = tile_d0 wide.
__device__ __forceinline__ void tdm_gather_rows(unsigned lds_addr, unsigned long long gaddr,
                                                unsigned dsz_code, unsigned tensor_d0,
                                                unsigned tensor_d1, unsigned tile_d0,
                                                unsigned nidx, unsigned long long stride0,
                                                const int* idx)
{
  u32x4 g0;
  g0[0] = 1u | (1u << 31);                      // gather_mode=1, 16-bit indices
  g0[1] = lds_addr;
  g0[2] = (unsigned)gaddr;
  g0[3] = ((unsigned)((gaddr >> 32) & 0x01FFFFFFull)) | (2u << 30);
  i32x8 g1;
  g1[0] = (int)(dsz_code << 16);
  g1[1] = (int)((tensor_d0 & 0xFFFFu) << 16);
  g1[2] = (int)(((tensor_d0 >> 16) & 0xFFFFu) | ((tensor_d1 & 0xFFFFu) << 16));
  g1[3] = (int)(((tensor_d1 >> 16) & 0xFFFFu) | (tile_d0 << 16));
  g1[4] = (int)(nidx & 0xFFFFu);                // tile_dim1 = #valid indices
  g1[5] = (int)(unsigned)(stride0 & 0xFFFFFFFFull);
  g1[6] = (int)(unsigned)((stride0 >> 32) & 0xFFFFull);
  g1[7] = 0;
  i32x4 g2, g3;
#pragma unroll
  for (int i = 0; i < 4; ++i){
    g2[i] = (idx[2*i]   & 0xFFFF) | ((idx[2*i+1]   & 0xFFFF) << 16);
    g3[i] = (idx[8+2*i] & 0xFFFF) | ((idx[8+2*i+1] & 0xFFFF) << 16);
  }
#if __clang_major__ >= 23
  i32x8 z8; z8[0]=0; z8[1]=0; z8[2]=0; z8[3]=0; z8[4]=0; z8[5]=0; z8[6]=0; z8[7]=0;
  __builtin_amdgcn_tensor_load_to_lds(g0, g1, g2, g3, z8, 0);
#else
  __builtin_amdgcn_tensor_load_to_lds(g0, g1, g2, g3, 0);
#endif
}

__device__ __forceinline__ unsigned lds_off(const void* p){
  return (unsigned)(size_t)p;   // generic LDS aperture: low 32 bits = LDS offset
}

// ---------------------------------------------------------------------------
// Prep: shuffle weights into per-lane-contiguous WMMA B-fragments (f16),
// compute centroid scalars (ru, sinh ru, cosh ru).
// ---------------------------------------------------------------------------
__global__ void prep_kernel(const float* __restrict__ Wih_f, const float* __restrict__ Wih_b,
                            const float* __restrict__ Whh_f, const float* __restrict__ Whh_b,
                            const float* __restrict__ attn_W, const float* __restrict__ centroid,
                            _Float16* __restrict__ wih_fr, _Float16* __restrict__ whh_fr,
                            _Float16* __restrict__ wat_fr, float* __restrict__ uscal)
{
  const int tid = blockIdx.x*blockDim.x + threadIdx.x;
  const int np  = gridDim.x*blockDim.x;

  for (int e = tid; e < 1536*EPAD; e += np){
    const int n = e / EPAD, k = e % EPAD;
    float v = 0.0f;
    if (k < EE) v = (n < G3) ? Wih_f[n*EE + k] : Wih_b[(n-G3)*EE + k];
    const int kc = k >> 5, kk = k & 31, nt = n >> 4;
    wih_fr[(((size_t)(kc*96 + nt))*32 + (((kk>>4)<<4) | (n&15)))*16 + (kk&15)] = (_Float16)v;
  }
  for (int e = tid; e < 2*G3*HH; e += np){
    const int dir = e / (G3*HH);
    const int r   = e % (G3*HH);
    const int n = r / HH, k = r % HH;
    const float v = dir ? Whh_b[n*HH + k] : Whh_f[n*HH + k];
    const int kc = k >> 5, kk = k & 31, nt = n >> 4;
    whh_fr[(size_t)dir*(384*512) +
           (((size_t)(kc*48 + nt))*32 + (((kk>>4)<<4) | (n&15)))*16 + (kk&15)] = (_Float16)v;
  }
  for (int e = tid; e < DD*DD; e += np){
    const int n = e / DD, k = e % DD;
    const float v = attn_W[n*DD + k];
    const int kc = k >> 5, kk = k & 31, nt = n >> 4;
    wat_fr[(((size_t)(kc*32 + nt))*32 + (((kk>>4)<<4) | (n&15)))*16 + (kk&15)] = (_Float16)v;
  }
  if (tid == 0){
    float s = 0.0f;
    for (int i = 0; i < DD; ++i){ float c = centroid[i]; s += c*c; }
    const float ru = sqrtf(s);
    uscal[0] = ru; uscal[1] = sinhf(ru); uscal[2] = coshf(ru);
  }
}

// ---------------------------------------------------------------------------
// xg = gather(emb, tokens) @ Wih_combined^T + bih -> (T*B, 1536) f16.
// Embedding gather via TDM gather-mode; per-N-tile B fragments preloaded into
// distinct registers so all 20 loads clause and overlap the wmma chain.
// ---------------------------------------------------------------------------
__global__ __launch_bounds__(256) void xg_proj_kernel(
    const int* __restrict__ tokens, const float* __restrict__ emb,
    const _Float16* __restrict__ wih_fr,
    const float* __restrict__ bih_f, const float* __restrict__ bih_b,
    _Float16* __restrict__ xg)
{
  __shared__ float    sgf[16*EE];     // TDM gather staging (f32)
  __shared__ _Float16 sx[16*328];
  const int mt = blockIdx.x;
  const int tid = threadIdx.x;
  const int lane = tid & 31, wave = tid >> 5;

  if (wave == 0){
    int idx[16];
#pragma unroll
    for (int i = 0; i < 16; ++i) idx[i] = tokens[mt*16 + i];  // uniform s_loads
    tdm_gather_rows(lds_off(sgf), (unsigned long long)(size_t)emb,
                    /*dsz=4B*/2u, /*td0*/EE, /*td1*/50000u,
                    /*tile_d0*/EE, /*nidx*/16u, /*stride0*/(unsigned long long)EE, idx);
    __builtin_amdgcn_s_wait_tensorcnt(0);
  }
  __syncthreads();
  for (int m = 0; m < 16; ++m)
    for (int k = tid; k < EPAD; k += 256)
      sx[m*328 + k] = (k < EE) ? (_Float16)sgf[m*EE + k] : (_Float16)0.0f;
  __syncthreads();

  // Hoist all 10 A-fragments (shared across N-tiles).
  v16h af[10];
#pragma unroll
  for (int kc = 0; kc < 10; ++kc) af[kc] = load_a_frag(sx, 328, kc, lane);

  const int mbase = (lane >> 4) * 8;
  for (int q = 0; q < 12; ++q){
    const int nt = wave*12 + q;
    const int n  = nt*16 + (lane & 15);
    const float bias = (n < G3) ? bih_f[n] : bih_b[n - G3];
    v16h bf[10];
#pragma unroll
    for (int kc = 0; kc < 10; ++kc) bf[kc] = load_b_frag(wih_fr, kc*96 + nt, lane);
    v8f c;
#pragma unroll
    for (int i = 0; i < 8; ++i) c[i] = bias;
#pragma unroll
    for (int kc = 0; kc < 10; ++kc) c = wmma_f16(af[kc], bf[kc], c);
#pragma unroll
    for (int v = 0; v < 8; ++v)
      xg[((size_t)(mt*16 + v + mbase))*1536 + n] = (_Float16)c[v];
  }
}

// ---------------------------------------------------------------------------
// GRU scan. Block = (dir, 16-batch-row tile); whole T-loop in-block.
// Whh B-fragments (48 x v16h) live in registers for the entire scan;
// A-fragment LDS loads pipelined one K-chunk ahead;
// per-step xg tile (16x768 f16) prefetched one step ahead via TDM.
// ---------------------------------------------------------------------------
__global__ __launch_bounds__(256) void gru_scan_kernel(
    const float* __restrict__ h0all, const _Float16* __restrict__ xg,
    const _Float16* __restrict__ whh_fr,
    const float* __restrict__ bhh_f, const float* __restrict__ bhh_b,
    float* __restrict__ fout)
{
  __shared__ _Float16 sh16[16*264];
  __shared__ float    sh32[16*260];
  __shared__ _Float16 sxg[16*G3];    // TDM-staged xg tile for current step
  const int blk = blockIdx.x;
  const int dir = blk >> 6;
  const int mt  = blk & 63;
  const int tid = threadIdx.x;
  const int lane = tid & 31, wave = tid >> 5;
  const float* bhh = dir ? bhh_b : bhh_f;
  const _Float16* wb = whh_fr + (size_t)dir * (384*512);

  for (int e = tid; e < 16*HH; e += 256){
    const int m = e >> 8, j = e & 255;
    const float h0 = h0all[(size_t)dir*BB*HH + (size_t)(mt*16 + m)*HH + j];
    sh32[m*260 + j] = h0;
    sh16[m*264 + j] = (_Float16)h0;
  }

  const int jc   = lane & 15;
  const int mrow = (lane >> 4) * 8;
  const int tr0  = 2*wave, tr1 = 2*wave + 1;
  const int ntg[6] = {tr0, 16+tr0, 32+tr0, tr1, 16+tr1, 32+tr1};

  // Hoist all Whh B-fragments for this wave's 6 gate tiles: read L2 once.
  v16h bfr[48];
#pragma unroll
  for (int kc = 0; kc < 8; ++kc)
#pragma unroll
    for (int g = 0; g < 6; ++g)
      bfr[kc*6 + g] = load_b_frag(wb, kc*48 + ntg[g], lane);

  // Kick off TDM for step 0's xg tile.
  if (wave == 0){
    const int t0 = dir ? (TT-1) : 0;
    const size_t elt = ((size_t)t0*BB + (size_t)mt*16)*1536 + (size_t)dir*G3;
    tdm_load_2d(lds_off(sxg), (unsigned long long)(size_t)(xg + elt),
                /*dsz=2B*/1u, 1536u, (unsigned)NROWS, (unsigned)G3, 16u, 1536ull);
  }
  __syncthreads();

  const float br0 = bhh[tr0*16 + jc], bz0 = bhh[256 + tr0*16 + jc], bn0 = bhh[512 + tr0*16 + jc];
  const float br1 = bhh[tr1*16 + jc], bz1 = bhh[256 + tr1*16 + jc], bn1 = bhh[512 + tr1*16 + jc];

  for (int s = 0; s < TT; ++s){
    const int t = dir ? (TT - 1 - s) : s;
    v8f cr[2], cz[2], cn[2];
#pragma unroll
    for (int i = 0; i < 8; ++i){
      cr[0][i] = br0; cz[0][i] = bz0; cn[0][i] = bn0;
      cr[1][i] = br1; cz[1][i] = bz1; cn[1][i] = bn1;
    }
    // Software-pipelined A loads: fetch kc+1 while wmma group kc executes.
    v16h a_cur = load_a_frag(sh16, 264, 0, lane);
#pragma unroll
    for (int kc = 0; kc < 8; ++kc){
      v16h a_nxt = a_cur;
      if (kc < 7) a_nxt = load_a_frag(sh16, 264, kc + 1, lane);
      cr[0] = wmma_f16(a_cur, bfr[kc*6+0], cr[0]);
      cz[0] = wmma_f16(a_cur, bfr[kc*6+1], cz[0]);
      cn[0] = wmma_f16(a_cur, bfr[kc*6+2], cn[0]);
      cr[1] = wmma_f16(a_cur, bfr[kc*6+3], cr[1]);
      cz[1] = wmma_f16(a_cur, bfr[kc*6+4], cz[1]);
      cn[1] = wmma_f16(a_cur, bfr[kc*6+5], cn[1]);
      a_cur = a_nxt;
    }
    if (wave == 0) __builtin_amdgcn_s_wait_tensorcnt(0);  // xg tile landed
    __syncthreads();

    const size_t rowbase = (size_t)t*BB + (size_t)mt*16;
#pragma unroll
    for (int u = 0; u < 2; ++u){
      const int j = (2*wave + u)*16 + jc;
#pragma unroll
      for (int v = 0; v < 8; ++v){
        const int m = v + mrow;
        const size_t row = rowbase + m;
        const float xr = (float)sxg[m*G3 + j];
        const float xz = (float)sxg[m*G3 + 256 + j];
        const float xn = (float)sxg[m*G3 + 512 + j];
        const float r  = sigm(xr + cr[u][v]);
        const float z  = sigm(xz + cz[u][v]);
        const float nn = tanhf(xn + r*cn[u][v]);
        const float h  = sh32[m*260 + j];
        const float hnew = (1.0f - z)*nn + z*h;
        sh32[m*260 + j] = hnew;
        sh16[m*264 + j] = (_Float16)hnew;
        fout[row*DD + dir*HH + j] = hnew;
      }
    }
    __syncthreads();   // h published; sxg fully consumed
    if (wave == 0 && s + 1 < TT){      // prefetch next step's xg during next GEMM
      const int tn = dir ? (TT - 2 - s) : (s + 1);
      const size_t elt = ((size_t)tn*BB + (size_t)mt*16)*1536 + (size_t)dir*G3;
      tdm_load_2d(lds_off(sxg), (unsigned long long)(size_t)(xg + elt),
                  1u, 1536u, (unsigned)NROWS, (unsigned)G3, 16u, 1536ull);
    }
  }
}

// ---------------------------------------------------------------------------
// Attention + hyperbolic scalars. f-tile staged via TDM 2D load; per-N-tile
// B fragments preloaded into registers for clause'd loads.
// ---------------------------------------------------------------------------
__global__ __launch_bounds__(256) void attn_kernel(
    const float* __restrict__ f, const _Float16* __restrict__ wat_fr,
    const float* __restrict__ attn_b, const float* __restrict__ centroid,
    const float* __restrict__ beta, const float* __restrict__ uscal,
    float* __restrict__ score, float* __restrict__ gam, float* __restrict__ scl)
{
  __shared__ float    sgf[16*DD];     // TDM staging (32 KB)
  __shared__ _Float16 sa[16*520];
  __shared__ float srr[16], sra[16], sdo[16];
  const int mt = blockIdx.x;
  const int tid = threadIdx.x;
  const int lane = tid & 31, wave = tid >> 5;

  if (tid < 16){ srr[tid] = 0.f; sra[tid] = 0.f; sdo[tid] = 0.f; }
  if (wave == 0){
    tdm_load_2d(lds_off(sgf), (unsigned long long)(size_t)(f + (size_t)mt*16*DD),
                /*dsz=4B*/2u, (unsigned)DD, (unsigned)NROWS, (unsigned)DD, 16u,
                (unsigned long long)DD);
    __builtin_amdgcn_s_wait_tensorcnt(0);
  }
  __syncthreads();
  for (int m = 0; m < 16; ++m){
    float part = 0.f;
    for (int k = tid; k < DD; k += 256){
      const float v = sgf[m*DD + k];
      sa[m*520 + k] = (_Float16)v;
      part += v*v;
    }
    atomicAdd(&srr[m], part);
  }
  __syncthreads();

  // Hoist all 16 A-fragments once (shared across this wave's 4 N-tiles).
  v16h af[16];
#pragma unroll
  for (int kc = 0; kc < 16; ++kc) af[kc] = load_a_frag(sa, 520, kc, lane);

  float pra[8], pdo[8];
#pragma unroll
  for (int v = 0; v < 8; ++v){ pra[v] = 0.f; pdo[v] = 0.f; }
  for (int q = 0; q < 4; ++q){
    const int nt = wave*4 + q;
    const int n  = nt*16 + (lane & 15);
    const float bias = attn_b[n];
    v16h bf[16];
#pragma unroll
    for (int kc = 0; kc < 16; ++kc) bf[kc] = load_b_frag(wat_fr, kc*32 + nt, lane);
    v8f c;
#pragma unroll
    for (int i = 0; i < 8; ++i) c[i] = bias;
#pragma unroll
    for (int kc = 0; kc < 16; ++kc) c = wmma_f16(af[kc], bf[kc], c);
    const float un = centroid[n];
#pragma unroll
    for (int v = 0; v < 8; ++v){
      const float av = tanhf(c[v]);
      pra[v] += av*av;
      pdo[v] += av*un;
    }
  }
  const int mb = (lane >> 4) * 8;
#pragma unroll
  for (int v = 0; v < 8; ++v){ atomicAdd(&sra[v+mb], pra[v]); atomicAdd(&sdo[v+mb], pdo[v]); }
  __syncthreads();

  if (tid < 16){
    const int m = tid;
    const size_t row = (size_t)mt*16 + m;
    const float rr = sqrtf(srr[m]);
    const float ra = sqrtf(sra[m]);
    const float ru = uscal[0], su = uscal[1], cu = uscal[2];
    float prod = coshf(ra)*cu - sinhf(ra)*su*(sdo[m]/(fmaxf(ra,1e-20f)*ru));
    prod = fminf(fmaxf(prod, 1.0f + EPS_), 1e16f);
    const float dist = logf(prod) + log1pf(sqrtf(prod*prod - 1.0f + EPS_)/prod);
    score[row] = -beta[0]*dist - 1.0f;
    const float th = tanhf(rr);
    float sech2 = fminf(fmaxf(1.0f - th*th, EPS_), 1.0f - EPS_);
    float g = 1.0f/sqrtf(sech2);
    gam[row] = fminf(fmaxf(g, 1.0f + EPS_), 1e16f);
    scl[row] = th/fmaxf(rr, 1e-20f);
  }
}

// ---------------------------------------------------------------------------
// Softmax-over-T pooling (gamma reweight + fK scale) and h_output copy.
// ---------------------------------------------------------------------------
__global__ __launch_bounds__(256) void pool_kernel(
    const float* __restrict__ f, const float* __restrict__ score,
    const float* __restrict__ gam, const float* __restrict__ scl,
    float* __restrict__ out)
{
  __shared__ float coef[TT];
  __shared__ float ssc[TT];
  const int b = blockIdx.x;
  const int tid = threadIdx.x;

  if (tid < TT) ssc[tid] = score[tid*BB + b];
  __syncthreads();
  if (tid == 0){
    float mx = -1e30f;
    for (int t = 0; t < TT; ++t) mx = fmaxf(mx, ssc[t]);
    float sum = 0.f;
    for (int t = 0; t < TT; ++t){
      const float e = __expf(ssc[t] - mx) * gam[t*BB + b];
      coef[t] = e; sum += e;
    }
    const float inv = 1.0f/sum;
    for (int t = 0; t < TT; ++t) coef[t] = coef[t]*inv*scl[t*BB + b];
  }
  __syncthreads();

  for (int d = tid; d < DD; d += 256){
    float acc = 0.f;
    for (int t = 0; t < TT; ++t)
      acc += coef[t] * f[((size_t)t*BB + b)*DD + d];
    out[(size_t)b*DD + d] = acc;
  }
  for (int j = tid; j < HH; j += 256){
    out[(size_t)BB*DD + (size_t)b*HH + j]                 = f[((size_t)(TT-1)*BB + b)*DD + j];
    out[(size_t)BB*DD + (size_t)BB*HH + (size_t)b*HH + j] = f[((size_t)b)*DD + HH + j];
  }
}

// ---------------------------------------------------------------------------
extern "C" void kernel_launch(void* const* d_in, const int* in_sizes, int n_in,
                              void* d_out, int out_size, void* d_ws, size_t ws_size,
                              hipStream_t stream)
{
  (void)in_sizes; (void)n_in; (void)out_size; (void)ws_size;
  const int*   tokens   = (const int*)  d_in[0];
  const float* hidden   = (const float*)d_in[1];
  const float* emb      = (const float*)d_in[2];
  const float* Wih_f    = (const float*)d_in[3];
  const float* Whh_f    = (const float*)d_in[4];
  const float* bih_f    = (const float*)d_in[5];
  const float* bhh_f    = (const float*)d_in[6];
  const float* Wih_b    = (const float*)d_in[7];
  const float* Whh_b    = (const float*)d_in[8];
  const float* bih_b    = (const float*)d_in[9];
  const float* bhh_b    = (const float*)d_in[10];
  const float* attn_W   = (const float*)d_in[11];
  const float* attn_b   = (const float*)d_in[12];
  const float* centroid = (const float*)d_in[13];
  const float* beta     = (const float*)d_in[14];
  float* out = (float*)d_out;

  char* w = (char*)d_ws;
  _Float16* ws_xg16  = (_Float16*)w; w += (size_t)NROWS*1536*sizeof(_Float16); // 192 MB
  float*    ws_f     = (float*)w;    w += (size_t)NROWS*DD*sizeof(float);      // 128 MB
  _Float16* ws_wih   = (_Float16*)w; w += (size_t)960*512*sizeof(_Float16);
  _Float16* ws_whh   = (_Float16*)w; w += (size_t)2*384*512*sizeof(_Float16);
  _Float16* ws_wat   = (_Float16*)w; w += (size_t)512*512*sizeof(_Float16);
  float*    ws_score = (float*)w;    w += (size_t)NROWS*sizeof(float);
  float*    ws_gamma = (float*)w;    w += (size_t)NROWS*sizeof(float);
  float*    ws_scale = (float*)w;    w += (size_t)NROWS*sizeof(float);
  float*    ws_uscal = (float*)w;    w += 16*sizeof(float);

  prep_kernel<<<dim3(1024), dim3(256), 0, stream>>>(
      Wih_f, Wih_b, Whh_f, Whh_b, attn_W, centroid, ws_wih, ws_whh, ws_wat, ws_uscal);
  xg_proj_kernel<<<dim3(4096), dim3(256), 0, stream>>>(
      tokens, emb, ws_wih, bih_f, bih_b, ws_xg16);
  gru_scan_kernel<<<dim3(128), dim3(256), 0, stream>>>(
      hidden, ws_xg16, ws_whh, bhh_f, bhh_b, ws_f);
  attn_kernel<<<dim3(4096), dim3(256), 0, stream>>>(
      ws_f, ws_wat, attn_b, centroid, beta, ws_uscal, ws_score, ws_gamma, ws_scale);
  pool_kernel<<<dim3(1024), dim3(256), 0, stream>>>(
      ws_f, ws_score, ws_gamma, ws_scale, out);
}
